// TrfEdgeNet_79645873537751
// MI455X (gfx1250) — matrix operand order repaired and checked by reading, not compiled, not run on hardware
//
#include <hip/hip_runtime.h>
#include <hip/hip_bf16.h>
#include <math.h>

#define NNODES 20000
#define NEDGES 100000
#define HC     1024     // HEADS*CH
#define CHW    128      // channels per head
#define NCLS   40

typedef unsigned short u16;
typedef __attribute__((ext_vector_type(16))) __bf16        v16bf;
typedef __attribute__((ext_vector_type(8)))  float         v8f;
typedef __attribute__((ext_vector_type(4)))  unsigned int  u32x4;
typedef __attribute__((ext_vector_type(2)))  unsigned int  u32x2;

union FragU { u32x4 u[2]; v16bf v; };

__device__ __forceinline__ u16 f2bf(float f) {
  union { float f; unsigned u; } c; c.f = f;
  unsigned r = c.u + 0x7fffu + ((c.u >> 16) & 1u);   // round-to-nearest-even
  return (u16)(r >> 16);
}
__device__ __forceinline__ float bf2f(unsigned bits16) {
  union { unsigned u; float f; } c; c.u = bits16 << 16;
  return c.f;
}
__device__ __forceinline__ void atomicMaxF(float* addr, float val) {
  if (val >= 0.0f) atomicMax((int*)addr, __float_as_int(val));
  else             atomicMin((unsigned int*)addr, __float_as_uint(val));
}

// ---------------------------------------------------------------------------
// WMMA GEMM: C[M,Nc] = A[M,K](bf16,row-major) * Bt[Nc,K](bf16, B transposed)
// One wave computes a 32x64 tile: 2 A-frags x 4 B-frags -> 8 accumulators.
// K is a compile-time constant -> full unroll; per k-step all 12 fragment
// loads are issued as one batch, then 8 WMMAs run back-to-back, letting the
// scheduler overlap next-step loads with current WMMAs.
// ---------------------------------------------------------------------------
template <int KC>
__global__ void __launch_bounds__(256)
gemm_bf16_wmma(const u16* __restrict__ A, const u16* __restrict__ Bt,
               const float* __restrict__ bias, void* __restrict__ Cout,
               int M, int Nc, int out_bf16)
{
  const int lane  = threadIdx.x & 31;
  const int wid   = (int)((blockIdx.x * blockDim.x + threadIdx.x) >> 5);
  const int ng    = Nc >> 6;
  const int total = (M >> 5) * ng;
  if (wid >= total) return;
  const int m0 = (wid / ng) << 5;
  const int n0 = (wid % ng) << 6;

  v8f acc[2][4] = {};
  const int l15 = lane & 15;
  const int kbA = (lane & 16) ? 8 : 0;    // A: elems 0..7 -> K=kbA+0..7, 8..15 -> K=kbA+16..23
  const int kbB = (lane & 16) ? 8 : 0;    // B: elems 0..7 -> K=kbB+0..7, 8..15 -> K=kbB+16..23

  const u16* arow0 = A + (size_t)(m0 + l15)      * KC + kbA;
  const u16* arow1 = A + (size_t)(m0 + 16 + l15) * KC + kbA;

#pragma unroll
  for (int k0 = 0; k0 < KC; k0 += 32) {
    FragU a[2], b[4];
    a[0].u[0] = *(const u32x4*)(arow0 + k0);
    a[0].u[1] = *(const u32x4*)(arow0 + k0 + 16);
    a[1].u[0] = *(const u32x4*)(arow1 + k0);
    a[1].u[1] = *(const u32x4*)(arow1 + k0 + 16);
#pragma unroll
    for (int t = 0; t < 4; ++t) {
      const u16* pb = Bt + (size_t)(n0 + t * 16 + l15) * KC + k0 + kbB;
      b[t].u[0] = *(const u32x4*)(pb);
      b[t].u[1] = *(const u32x4*)(pb + 16);
    }
#pragma unroll
    for (int t = 0; t < 4; ++t) {
      acc[0][t] = __builtin_amdgcn_wmma_f32_16x16x32_bf16(
                    false, a[0].v, false, b[t].v, (short)0, acc[0][t], false, false);
      acc[1][t] = __builtin_amdgcn_wmma_f32_16x16x32_bf16(
                    false, a[1].v, false, b[t].v, (short)0, acc[1][t], false, false);
    }
  }

#pragma unroll
  for (int mt = 0; mt < 2; ++mt) {
    const int mrow = m0 + mt * 16 + ((lane & 16) ? 8 : 0);
#pragma unroll
    for (int t = 0; t < 4; ++t) {
      const int nb = n0 + t * 16 + l15;
      const float bv = bias ? bias[nb] : 0.0f;
#pragma unroll
      for (int r = 0; r < 8; ++r) {
        const float val = acc[mt][t][r] + bv;
        const size_t idx = (size_t)(mrow + r) * (size_t)Nc + (size_t)nb;
        if (out_bf16) ((u16*)Cout)[idx]   = f2bf(val);
        else          ((float*)Cout)[idx] = val;
      }
    }
  }
}

// ---------------------------------------------------------------------------
// small utility kernels
// ---------------------------------------------------------------------------
__global__ void cvt_bf16_kernel(const float* __restrict__ in, u16* __restrict__ out, int n) {
  int t = blockIdx.x * blockDim.x + threadIdx.x;
  if (t < n) out[t] = f2bf(in[t]);
}

// Wt[n*Kpad + kp] = (kp<K) ? W[kp*Nc + n] : 0   (bf16)
__global__ void cvt_transpose_kernel(const float* __restrict__ W, u16* __restrict__ Wt,
                                     int K, int Nc, int Kpad) {
  int t = blockIdx.x * blockDim.x + threadIdx.x;
  if (t >= Nc * Kpad) return;
  int n = t / Kpad, kp = t - n * Kpad;
  Wt[t] = (kp < K) ? f2bf(W[(size_t)kp * Nc + n]) : (u16)0;
}

__global__ void pad_edge_attr_kernel(const float* __restrict__ ea, u16* __restrict__ out, int E) {
  int t = blockIdx.x * blockDim.x + threadIdx.x;
  if (t >= E * 32) return;
  int e = t >> 5, k = t & 31;
  out[t] = (k < 16) ? f2bf(ea[e * 16 + k]) : (u16)0;
}

__global__ void fill_kernel(float* __restrict__ p, float v, int n) {
  int t = blockIdx.x * blockDim.x + threadIdx.x;
  if (t < n) p[t] = v;
}

// ---------------------------------------------------------------------------
// edge kernels: one wave32 per edge; lane covers 4 contiguous channels.
// ---------------------------------------------------------------------------
__global__ void __launch_bounds__(256)
edge_alpha_kernel(const int* __restrict__ src, const int* __restrict__ dst,
                  const u16* __restrict__ Q, const u16* __restrict__ Kb,
                  const u16* __restrict__ Eb, float* __restrict__ alpha,
                  float* __restrict__ amax, int E)
{
  const int wid = (int)((blockIdx.x * blockDim.x + threadIdx.x) >> 5);
  if (wid >= E) return;
  const int lane = threadIdx.x & 31;
  const int s = src[wid], d = dst[wid];
  const u16* qrow = Q  + (size_t)d   * HC;
  const u16* krow = Kb + (size_t)s   * HC;
  const u16* erow = Eb + (size_t)wid * HC;
#pragma unroll
  for (int h = 0; h < 8; ++h) {
    const int c = h * CHW + lane * 4;
    u32x2 qu = *(const u32x2*)(qrow + c);
    u32x2 ku = *(const u32x2*)(krow + c);
    u32x2 eu = *(const u32x2*)(erow + c);
    float acc = 0.0f;
    acc += bf2f(qu[0] & 0xffffu) * (bf2f(ku[0] & 0xffffu) + bf2f(eu[0] & 0xffffu));
    acc += bf2f(qu[0] >> 16)     * (bf2f(ku[0] >> 16)     + bf2f(eu[0] >> 16));
    acc += bf2f(qu[1] & 0xffffu) * (bf2f(ku[1] & 0xffffu) + bf2f(eu[1] & 0xffffu));
    acc += bf2f(qu[1] >> 16)     * (bf2f(ku[1] >> 16)     + bf2f(eu[1] >> 16));
#pragma unroll
    for (int m = 16; m >= 1; m >>= 1) acc += __shfl_xor(acc, m, 32);
    if (lane == 0) {
      const float a = acc * 0.088388347648318447f;   // 1/sqrt(128)
      alpha[(size_t)wid * 8 + h] = a;
      atomicMaxF(&amax[(size_t)d * 8 + h], a);
    }
  }
}

__global__ void edge_softmax_kernel(const int* __restrict__ dst,
                                    float* __restrict__ alpha,
                                    const float* __restrict__ amax,
                                    float* __restrict__ denom, int E)
{
  int t = blockIdx.x * blockDim.x + threadIdx.x;
  if (t >= E * 8) return;
  int e = t >> 3, h = t & 7;
  int d = dst[e];
  float ex = __expf(alpha[t] - amax[(size_t)d * 8 + h]);
  alpha[t] = ex;
  atomicAdd(&denom[(size_t)d * 8 + h], ex);
}

__global__ void __launch_bounds__(256)
edge_scatter_kernel(const int* __restrict__ src, const int* __restrict__ dst,
                    const u16* __restrict__ Vb, const u16* __restrict__ Eb,
                    const float* __restrict__ alpha, const float* __restrict__ denom,
                    float* __restrict__ agg, int E)
{
  const int wid = (int)((blockIdx.x * blockDim.x + threadIdx.x) >> 5);
  if (wid >= E) return;
  const int lane = threadIdx.x & 31;
  const int s = src[wid], d = dst[wid];
  const u16* vrow = Vb + (size_t)s   * HC;
  const u16* erow = Eb + (size_t)wid * HC;
#pragma unroll
  for (int h = 0; h < 8; ++h) {
    const float aw = alpha[(size_t)wid * 8 + h] / denom[(size_t)d * 8 + h];
    const int c = h * CHW + lane * 4;
    u32x2 vu = *(const u32x2*)(vrow + c);
    u32x2 eu = *(const u32x2*)(erow + c);
    float* ap = agg + (size_t)d * HC + c;
    atomicAdd(ap + 0, (bf2f(vu[0] & 0xffffu) + bf2f(eu[0] & 0xffffu)) * aw);
    atomicAdd(ap + 1, (bf2f(vu[0] >> 16)     + bf2f(eu[0] >> 16))     * aw);
    atomicAdd(ap + 2, (bf2f(vu[1] & 0xffffu) + bf2f(eu[1] & 0xffffu)) * aw);
    atomicAdd(ap + 3, (bf2f(vu[1] >> 16)     + bf2f(eu[1] >> 16))     * aw);
  }
}

// head mean + root skip + (ReLU | BatchNorm)
__global__ void finalize_kernel(const float* __restrict__ agg, const float* __restrict__ skip,
                                float* __restrict__ hout, int mode,
                                const float* __restrict__ gamma, const float* __restrict__ beta,
                                const float* __restrict__ mean,  const float* __restrict__ var)
{
  int t = blockIdx.x * blockDim.x + threadIdx.x;
  if (t >= NNODES * CHW) return;
  int n = t >> 7, c = t & 127;
  float s = 0.0f;
#pragma unroll
  for (int h = 0; h < 8; ++h) s += agg[(size_t)n * HC + h * CHW + c];
  s = s * 0.125f + skip[t];
  if (mode == 0) s = fmaxf(s, 0.0f);
  else s = (s - mean[c]) * rsqrtf(var[c] + 1e-5f) * gamma[c] + beta[c];
  hout[t] = s;
}

__global__ void classifier_kernel(const float* __restrict__ h, const float* __restrict__ W,
                                  const float* __restrict__ b, float* __restrict__ out)
{
  __shared__ float row[CHW];
  const int n = blockIdx.x, t = threadIdx.x;
  row[t] = h[(size_t)n * CHW + t];
  __syncthreads();
  if (t < NCLS) {
    float s = b[t];
#pragma unroll
    for (int c = 0; c < CHW; ++c) s += row[c] * W[c * NCLS + t];
    out[(size_t)n * NCLS + t] = s;
  }
}

// ---------------------------------------------------------------------------
extern "C" void kernel_launch(void* const* d_in, const int* in_sizes, int n_in,
                              void* d_out, int out_size, void* d_ws, size_t ws_size,
                              hipStream_t stream)
{
  (void)in_sizes; (void)n_in; (void)out_size; (void)ws_size;
  const int N = NNODES, E = NEDGES;

  const float* x   = (const float*)d_in[0];
  const int*   ei  = (const int*)  d_in[1];
  const float* ea  = (const float*)d_in[2];
  const int* src = ei;
  const int* dst = ei + E;
  const float* gamma = (const float*)d_in[30];
  const float* beta  = (const float*)d_in[31];
  const float* mean  = (const float*)d_in[32];
  const float* var   = (const float*)d_in[33];
  const float* Wcls  = (const float*)d_in[34];
  const float* bcls  = (const float*)d_in[35];

  // ---- bump allocator over d_ws ----
  size_t off = 0; char* base = (char*)d_ws;
  auto take = [&](size_t bytes) -> void* {
    void* p = base + off;
    off = (off + bytes + 255) & ~(size_t)255;
    return p;
  };
  u16* hb   = (u16*)take((size_t)N * CHW * 2);
  u16* eap  = (u16*)take((size_t)E * 32 * 2);
  u16* Wtq  = (u16*)take((size_t)HC * CHW * 2);
  u16* Wtk  = (u16*)take((size_t)HC * CHW * 2);
  u16* Wtv  = (u16*)take((size_t)HC * CHW * 2);
  u16* Wts  = (u16*)take((size_t)CHW * CHW * 2);
  u16* Wte  = (u16*)take((size_t)HC * 32 * 2);
  u16* Qb   = (u16*)take((size_t)N * HC * 2);
  u16* Kb   = (u16*)take((size_t)N * HC * 2);
  u16* Vb   = (u16*)take((size_t)N * HC * 2);
  u16* Ebuf = (u16*)take((size_t)E * HC * 2);
  float* skip  = (float*)take((size_t)N * CHW * 4);
  float* alpha = (float*)take((size_t)E * 8 * 4);
  float* amax  = (float*)take((size_t)N * 8 * 4);
  float* denom = (float*)take((size_t)N * 8 * 4);
  float* agg   = (float*)take((size_t)N * HC * 4);
  float* h0    = (float*)take((size_t)N * CHW * 4);
  float* h1    = (float*)take((size_t)N * CHW * 4);

  auto nb = [](long long th) { return (unsigned)((th + 255) / 256); };
  // tiles are 32x64 per wave
  auto gemm128 = [&](const u16* A, const u16* Bt, const float* bias, void* C,
                     int M, int Nc, int outbf) {
    long long waves = (long long)(M / 32) * (Nc / 64);
    gemm_bf16_wmma<128><<<nb(waves * 32), 256, 0, stream>>>(A, Bt, bias, C, M, Nc, outbf);
  };
  auto gemm32 = [&](const u16* A, const u16* Bt, const float* bias, void* C,
                    int M, int Nc, int outbf) {
    long long waves = (long long)(M / 32) * (Nc / 64);
    gemm_bf16_wmma<32><<<nb(waves * 32), 256, 0, stream>>>(A, Bt, bias, C, M, Nc, outbf);
  };

  // edge_attr padded to K=32, once (re-used by all layers)
  pad_edge_attr_kernel<<<nb((long long)E * 32), 256, 0, stream>>>(ea, eap, E);

  const float* hin = x;
  float* houts[3] = { h0, h1, h0 };

  for (int L = 0; L < 3; ++L) {
    const float* Wq = (const float*)d_in[3 + 9 * L + 0];
    const float* bq = (const float*)d_in[3 + 9 * L + 1];
    const float* Wk = (const float*)d_in[3 + 9 * L + 2];
    const float* bk = (const float*)d_in[3 + 9 * L + 3];
    const float* Wv = (const float*)d_in[3 + 9 * L + 4];
    const float* bv = (const float*)d_in[3 + 9 * L + 5];
    const float* We = (const float*)d_in[3 + 9 * L + 6];
    const float* Ws = (const float*)d_in[3 + 9 * L + 7];
    const float* bs = (const float*)d_in[3 + 9 * L + 8];

    // bf16 conversions (input features + transposed weights)
    cvt_bf16_kernel<<<nb((long long)N * CHW), 256, 0, stream>>>(hin, hb, N * CHW);
    cvt_transpose_kernel<<<nb((long long)HC * CHW), 256, 0, stream>>>(Wq, Wtq, CHW, HC, CHW);
    cvt_transpose_kernel<<<nb((long long)HC * CHW), 256, 0, stream>>>(Wk, Wtk, CHW, HC, CHW);
    cvt_transpose_kernel<<<nb((long long)HC * CHW), 256, 0, stream>>>(Wv, Wtv, CHW, HC, CHW);
    cvt_transpose_kernel<<<nb((long long)CHW * CHW), 256, 0, stream>>>(Ws, Wts, CHW, CHW, CHW);
    cvt_transpose_kernel<<<nb((long long)HC * 32), 256, 0, stream>>>(We, Wte, 16, HC, 32);

    // WMMA GEMMs: Q/K/V (bf16 out), skip (f32 out), edge embedding (bf16 out)
    gemm128(hb,  Wtq, bq, Qb,   N, HC,  1);
    gemm128(hb,  Wtk, bk, Kb,   N, HC,  1);
    gemm128(hb,  Wtv, bv, Vb,   N, HC,  1);
    gemm128(hb,  Wts, bs, skip, N, CHW, 0);
    gemm32 (eap, Wte, nullptr, Ebuf, E, HC, 1);

    // segment softmax state
    fill_kernel<<<nb((long long)N * 8), 256, 0, stream>>>(amax, -INFINITY, N * 8);
    fill_kernel<<<nb((long long)N * 8), 256, 0, stream>>>(denom, 0.0f, N * 8);
    fill_kernel<<<nb((long long)N * HC), 256, 0, stream>>>(agg, 0.0f, N * HC);

    edge_alpha_kernel<<<nb((long long)E * 32), 256, 0, stream>>>(src, dst, Qb, Kb, Ebuf,
                                                                 alpha, amax, E);
    edge_softmax_kernel<<<nb((long long)E * 8), 256, 0, stream>>>(dst, alpha, amax, denom, E);
    edge_scatter_kernel<<<nb((long long)E * 32), 256, 0, stream>>>(src, dst, Vb, Ebuf,
                                                                   alpha, denom, agg, E);

    finalize_kernel<<<nb((long long)N * CHW), 256, 0, stream>>>(
        agg, skip, houts[L], (L < 2) ? 0 : 1, gamma, beta, mean, var);
    hin = houts[L];
  }

  classifier_kernel<<<N, CHW, 0, stream>>>(hin, Wcls, bcls, (float*)d_out);
}
